// Net_51419348467766
// MI455X (gfx1250) — matrix operand order repaired and checked by reading, not compiled, not run on hardware
//
#include <hip/hip_runtime.h>
#include <cstdint>

#define UNITS 64
#define GN_EPS 1e-5f

typedef __attribute__((ext_vector_type(16))) __bf16 v16bf;
typedef __attribute__((ext_vector_type(8)))  float  v8f;
typedef __attribute__((ext_vector_type(4)))  unsigned int v4u;
typedef __attribute__((ext_vector_type(4)))  float v4f;

__device__ __forceinline__ unsigned short f2bf(float x) {
  unsigned int u = __float_as_uint(x);
  u += 0x7fffu + ((u >> 16) & 1u);          // round-to-nearest-even
  return (unsigned short)(u >> 16);
}

union Frag {            // one WMMA 16x32 bf16 operand = 8 VGPRs per lane
  v16bf v;
  v4u   q[2];
};

// ---------------------------------------------------------------------------
// Fused 64-wide GEMM:  out = A(bf16)[n,64] @ W(f32->bf16)[64,64]  (+ epilogue)
// mode 0 (conv):  outF = acc (raw t);  outP = acc*dinv[row]^2 + bias[col]
// mode 1 (head):  g = relu(res + acc + bias); outF = g (f32); outB = bf16(g)
// 128 threads = 4 waves; each wave owns a 16-row stripe, 4 x (16x16) C tiles.
// ---------------------------------------------------------------------------
__global__ __launch_bounds__(128) void gemm64_kernel(
    const unsigned short* __restrict__ A,
    const float* __restrict__ W,
    const float* __restrict__ bias,
    const float* __restrict__ dinv,
    const float* __restrict__ res,
    float* __restrict__ outF,
    float* __restrict__ outP,
    unsigned short* __restrict__ outB,
    int n, int mode)
{
  __shared__ __align__(16) unsigned short Wt[UNITS * UNITS]; // W transposed, bf16: Wt[n][k]
  for (int idx = threadIdx.x; idx < UNITS * UNITS; idx += 128) {
    int k = idx >> 6, c = idx & 63;
    Wt[c * UNITS + k] = f2bf(W[idx]);
  }
  __syncthreads();

  const int wave = threadIdx.x >> 5;
  const int lane = threadIdx.x & 31;
  const int half = lane >> 4;     // lane group 0..15 / 16..31
  const int lm   = lane & 15;
  const int row0 = (blockIdx.x * 4 + wave) * 16;

  // A fragments: lane holds row (row0+lm); element e -> K = k0 + half*8 + e (e<8)
  //                                                    K = k0 + 8 + half*8 + e (e>=8)
  int rA = row0 + lm;
  if (rA > n - 1) rA = n - 1;                 // clamp: results discarded at store
  const unsigned short* arow = A + (size_t)rA * UNITS;

  Frag a0, a1;
  a0.q[0] = *(const v4u*)(arow + half * 8);        // K =  0..31 block
  a0.q[1] = *(const v4u*)(arow + 16 + half * 8);
  a1.q[0] = *(const v4u*)(arow + 32 + half * 8);   // K = 32..63 block
  a1.q[1] = *(const v4u*)(arow + 48 + half * 8);

  v8f acc[4] = {};
  #pragma unroll
  for (int t = 0; t < 4; ++t) {
    // B fragment: lane holds col (t*16+lm); element e -> K = k0 + half*8 + e (e<8), k0 + 8 + half*8 + e (e>=8)
    const unsigned short* wp = &Wt[(t * 16 + lm) * UNITS + half * 8];
    Frag b0, b1;
    b0.q[0] = *(const v4u*)(wp);
    b0.q[1] = *(const v4u*)(wp + 16);
    b1.q[0] = *(const v4u*)(wp + 32);
    b1.q[1] = *(const v4u*)(wp + 48);
    acc[t] = __builtin_amdgcn_wmma_f32_16x16x32_bf16(false, a0.v, false, b0.v,
                                                     (short)0, acc[t], false, false);
    acc[t] = __builtin_amdgcn_wmma_f32_16x16x32_bf16(false, a1.v, false, b1.v,
                                                     (short)0, acc[t], false, false);
  }

  // C/D layout: VGPR r, lane-half h -> row (r + h*8), col lm (within tile)
  #pragma unroll
  for (int t = 0; t < 4; ++t) {
    const int col = t * 16 + lm;
    #pragma unroll
    for (int r = 0; r < 8; ++r) {
      const int grow = row0 + half * 8 + r;
      if (grow < n) {
        const float v = acc[t][r];
        const size_t off = (size_t)grow * UNITS + col;
        if (mode == 0) {
          outF[off] = v;
          const float dn = dinv[grow];
          outP[off] = v * dn * dn + bias[col];
        } else {
          float g = res[off] + v + bias[col];
          g = g > 0.f ? g : 0.f;
          outF[off] = g;
          outB[off] = f2bf(g);
        }
      }
    }
  }
}

// ---------------------------------------------------------------------------
__global__ void dinv_init_kernel(float* dinv, int n) {
  int i = blockIdx.x * blockDim.x + threadIdx.x;
  if (i < n) dinv[i] = 1.0f;                    // self-loop contributes 1
}

__global__ void deg_kernel(const int* __restrict__ ei, float* dinv, int nE) {
  int e = blockIdx.x * blockDim.x + threadIdx.x;
  if (e < nE) atomicAdd(&dinv[ei[nE + e]], 1.0f);   // count dst occurrences
}

__global__ void dinv_fin_kernel(float* dinv, int n) {
  int i = blockIdx.x * blockDim.x + threadIdx.x;
  if (i < n) dinv[i] = rsqrtf(dinv[i]);
}

// layer 0: t = x (N,1) @ w0 (1,64); also self-loop pre-accumulate
__global__ void layer0_kernel(const float* __restrict__ x, const float* __restrict__ w0,
                              const float* __restrict__ b0, const float* __restrict__ dinv,
                              float* __restrict__ outT, float* __restrict__ outP, int n) {
  int idx = blockIdx.x * blockDim.x + threadIdx.x;
  if (idx >= n * UNITS) return;
  int r = idx >> 6, c = idx & 63;
  float t = x[r] * w0[c];
  outT[idx] = t;
  float dn = dinv[r];
  outP[idx] = t * dn * dn + b0[c];
}

// scatter-add over edges: pre[dst] += t[src] * dinv[src] * dinv[dst]
// 16 threads per edge, 4 channels each (vectorized gather, f32 atomics in L2)
__global__ void aggregate_kernel(const float* __restrict__ t, const int* __restrict__ ei,
                                 const float* __restrict__ dinv, float* __restrict__ pre,
                                 int nE) {
  int idx = blockIdx.x * blockDim.x + threadIdx.x;
  int e = idx >> 4;
  if (e >= nE) return;
  int c4 = (idx & 15) * 4;
  int s = ei[e], d = ei[nE + e];
  float w = dinv[s] * dinv[d];
  v4f tv = *(const v4f*)(t + (size_t)s * UNITS + c4);
  float* p = pre + (size_t)d * UNITS + c4;
  atomicAdd(p + 0, tv.x * w);
  atomicAdd(p + 1, tv.y * w);
  atomicAdd(p + 2, tv.z * w);
  atomicAdd(p + 3, tv.w * w);
}

__global__ void zero_stats_kernel(float* s) { s[threadIdx.x] = 0.f; }  // 128 floats

// per-channel sum and sum-of-squares over all nodes
__global__ void gn_stats_kernel(const float* __restrict__ h, float* __restrict__ s, int n) {
  int c = threadIdx.x & 63;
  int rpb = blockDim.x >> 6;
  int r = blockIdx.x * rpb + (threadIdx.x >> 6);
  int stride = gridDim.x * rpb;
  float a = 0.f, b = 0.f;
  for (; r < n; r += stride) {
    float v = h[(size_t)r * UNITS + c];
    a += v; b += v * v;
  }
  atomicAdd(&s[c], a);
  atomicAdd(&s[UNITS + c], b);
}

// GraphNorm + SiLU, fused with bf16 conversion for the next GEMM's A operand
__global__ void gn_apply_kernel(const float* __restrict__ h, const float* __restrict__ s,
                                const float* __restrict__ gamma, const float* __restrict__ beta,
                                const float* __restrict__ alpha,
                                unsigned short* __restrict__ hbOut, int n) {
  int idx = blockIdx.x * blockDim.x + threadIdx.x;
  if (idx >= n * UNITS) return;
  int c = idx & 63;
  float inv_n = 1.0f / (float)n;
  float mu = s[c] * inv_n;
  float al = alpha[c];
  // E[(h - a*mu)^2] = E[h^2] - 2a*mu^2 + a^2*mu^2
  float var = s[UNITS + c] * inv_n - 2.f * al * mu * mu + al * al * mu * mu;
  float hc = h[idx] - al * mu;
  float y = gamma[c] * hc * rsqrtf(var + GN_EPS) + beta[c];
  y = y / (1.f + expf(-y));   // SiLU
  hbOut[idx] = f2bf(y);
}

__global__ void cvt_kernel(const float* __restrict__ in, unsigned short* __restrict__ out,
                           int n64) {
  int idx = blockIdx.x * blockDim.x + threadIdx.x;
  if (idx < n64) out[idx] = f2bf(in[idx]);
}

// final 64->1 head: sigmoid(g . wl + bl)
__global__ void head_kernel(const float* __restrict__ g, const float* __restrict__ wl,
                            const float* __restrict__ bl, float* __restrict__ out, int n) {
  int r = blockIdx.x * blockDim.x + threadIdx.x;
  if (r >= n) return;
  float acc = bl[0];
  const v4f* gp = (const v4f*)(g + (size_t)r * UNITS);
  #pragma unroll
  for (int i = 0; i < 16; ++i) {
    v4f a = gp[i];
    v4f b = *(const v4f*)(wl + i * 4);
    acc += a.x * b.x + a.y * b.y + a.z * b.z + a.w * b.w;
  }
  out[r] = 1.f / (1.f + expf(-acc));
}

// ---------------------------------------------------------------------------
extern "C" void kernel_launch(void* const* d_in, const int* in_sizes, int n_in,
                              void* d_out, int out_size, void* d_ws, size_t ws_size,
                              hipStream_t stream)
{
  const float* x    = (const float*)d_in[0];
  const int*   ei   = (const int*)  d_in[1];
  const float* cw0  = (const float*)d_in[2];
  const float* cw   = (const float*)d_in[3];
  const float* cb   = (const float*)d_in[4];
  const float* gng  = (const float*)d_in[5];
  const float* gnb  = (const float*)d_in[6];
  const float* gna  = (const float*)d_in[7];
  const float* pw   = (const float*)d_in[8];
  const float* pb   = (const float*)d_in[9];
  const float* pwl  = (const float*)d_in[10];
  const float* pbl  = (const float*)d_in[11];
  const float* hw   = (const float*)d_in[12];
  const float* hbb  = (const float*)d_in[13];
  const float* hwl  = (const float*)d_in[14];
  const float* hbl  = (const float*)d_in[15];

  const int n  = in_sizes[0];       // N_NODES  (x is [N,1])
  const int nE = in_sizes[1] / 2;   // edge_index is [2,E]

  // workspace layout (all 256B-aligned since 64*4 = 256)
  const size_t per = (size_t)n * UNITS;
  float* bufT = (float*)d_ws;                       // raw t / head ping
  float* bufP = bufT + per;                         // pre-agg / final embedding
  float* bufG = bufP + per;                         // head pong
  unsigned short* hbuf = (unsigned short*)(bufG + per);  // bf16 GEMM A operand
  float* dinv  = (float*)(hbuf + per);
  float* stats = dinv + n;                          // 128 floats: S1 | S2

  const int T = 256;
  const int gN    = (n + T - 1) / T;
  const int gNC   = (int)((per + T - 1) / T);
  const int gE    = (nE + T - 1) / T;
  const int gE16  = (int)(((size_t)nE * 16 + T - 1) / T);
  const int gGemm = (n + 63) / 64;

  // GCN normalization: dinv = rsqrt(1 + deg)
  dinv_init_kernel<<<gN, T, 0, stream>>>(dinv, n);
  deg_kernel<<<gE, T, 0, stream>>>(ei, dinv, nE);
  dinv_fin_kernel<<<gN, T, 0, stream>>>(dinv, n);

  // 12 conv layers
  for (int i = 0; i < 12; ++i) {
    if (i == 0) {
      layer0_kernel<<<gNC, T, 0, stream>>>(x, cw0, cb, dinv, bufT, bufP, n);
    } else {
      gemm64_kernel<<<gGemm, 128, 0, stream>>>(
          hbuf, cw + (size_t)(i - 1) * UNITS * UNITS, cb + (size_t)i * UNITS,
          dinv, nullptr, bufT, bufP, nullptr, n, 0);
    }
    aggregate_kernel<<<gE16, T, 0, stream>>>(bufT, ei, dinv, bufP, nE);
    if (i < 11) {
      zero_stats_kernel<<<1, 128, 0, stream>>>(stats);
      gn_stats_kernel<<<1024, T, 0, stream>>>(bufP, stats, n);
      gn_apply_kernel<<<gNC, T, 0, stream>>>(
          bufP, stats, gng + (size_t)i * UNITS, gnb + (size_t)i * UNITS,
          gna + (size_t)i * UNITS, hbuf, n);
    }
  }
  // bufP = final embedding h (preserved through both heads)

  float* dout = (float*)d_out;
  for (int head = 0; head < 2; ++head) {
    const float* W  = head ? hw  : pw;
    const float* B  = head ? hbb : pb;
    const float* WL = head ? hwl : pwl;
    const float* BL = head ? hbl : pbl;

    cvt_kernel<<<gNC, T, 0, stream>>>(bufP, hbuf, (int)per);
    const float* resP[4] = { bufP, bufG, bufT, bufG };
    float*       outP[4] = { bufG, bufT, bufG, bufT };
    for (int l = 0; l < 4; ++l) {
      gemm64_kernel<<<gGemm, 128, 0, stream>>>(
          hbuf, W + (size_t)l * UNITS * UNITS, B + (size_t)l * UNITS,
          nullptr, resP[l], outP[l], nullptr, hbuf, n, 1);
    }
    head_kernel<<<gN, T, 0, stream>>>(bufT, WL, BL, dout + (size_t)head * n, n);
  }
}
